// SwitchDropTokenCalculator_74380243632186
// MI455X (gfx1250) — matrix operand order repaired and checked by reading, not compiled, not run on hardware
//
#include <hip/hip_runtime.h>
#include <math.h>

// ---- problem constants (from reference) ----
#define NEXP 8
#define NTOK 8192
#define DH   2048
#define FF   1376
#define CAP  1280          // int(1.25 * 8192 / 8)
#define SCORE_SCALE 1.0f

typedef __attribute__((ext_vector_type(8)))  float  v8f;
typedef __attribute__((ext_vector_type(16))) __bf16 v16bf;

static __device__ __forceinline__ v8f wmma_bf16(v16bf a, v16bf b, v8f c) {
  // D(16x16 f32) = A(16x32 bf16) * B(32x16 bf16) + C
  return __builtin_amdgcn_wmma_f32_16x16x32_bf16(
      /*neg_a=*/false, a, /*neg_b=*/false, b,
      /*c_mod=*/(short)0, c, /*reuse_a=*/false, /*reuse_b=*/false);
}

static __device__ __forceinline__ unsigned short f2bf(float f) {
  unsigned u = __float_as_uint(f);
  unsigned r = u + 0x7FFFu + ((u >> 16) & 1u);   // round-to-nearest-even
  return (unsigned short)(r >> 16);
}
static __device__ __forceinline__ float bf2f(unsigned short b) {
  return __uint_as_float(((unsigned)b) << 16);
}

union Frag { v16bf v; uint4 q[2]; };

// A fragment (16x32 bf16): per-lane K = 8h..8h+7 then 16+8h..16+8h+7 (h = lane/16), dword offsets sel..sel+3 and sel+8..sel+11
static __device__ __forceinline__ v16bf load_a_frag(const unsigned int* p, int sel) {
  Frag f;
  f.q[0] = *(const uint4*)(p + sel);
  f.q[1] = *(const uint4*)(p + sel + 8);
  return f.v;
}
// B fragment (32x16 bf16): per-lane K = 8h..8h+7 then 16+8h..16+8h+7 (h = lane/16), dword offsets sel..sel+3 and sel+8..sel+11
static __device__ __forceinline__ v16bf load_b_frag(const unsigned int* p, int sel) {
  Frag f;
  f.q[0] = *(const uint4*)(p + sel);
  f.q[1] = *(const uint4*)(p + sel + 8);
  return f.v;
}

// ---------------------------------------------------------------------------
// Routing: deterministic rank = #earlier tokens with same expert.
// ---------------------------------------------------------------------------
__global__ __launch_bounds__(256) void moe_route(const int* __restrict__ topk_idx,
                                                 int* __restrict__ token_of_slot) {
  __shared__ int base[NEXP];
  __shared__ int sh_e[256];
  const int tid = threadIdx.x;

  for (int i = tid; i < NEXP * CAP; i += 256) token_of_slot[i] = -1;
  if (tid < NEXP) base[tid] = 0;
  __syncthreads();

  for (int c = 0; c < NTOK / 256; ++c) {
    const int t = c * 256 + tid;
    const int e = topk_idx[t];
    sh_e[tid] = e;
    __syncthreads();
    int r = base[e];
    for (int j = 0; j < tid; ++j) r += (sh_e[j] == e) ? 1 : 0;
    if (r < CAP) token_of_slot[e * CAP + r] = t;
    __syncthreads();
    if (tid < NEXP) {
      int cnt = 0;
      for (int j = 0; j < 256; ++j) cnt += (sh_e[j] == tid) ? 1 : 0;
      base[tid] += cnt;
    }
    __syncthreads();
  }
}

// ---------------------------------------------------------------------------
// Zero output (dropped tokens must be exactly zero; harness poisons d_out).
// ---------------------------------------------------------------------------
__global__ __launch_bounds__(256) void moe_zero_out(float4* __restrict__ out) {
  const size_t i = (size_t)blockIdx.x * blockDim.x + threadIdx.x;
  out[i] = make_float4(0.f, 0.f, 0.f, 0.f);
}

// ---------------------------------------------------------------------------
// Split fp32 -> (hi, lo) bf16 planes, same layout. 4 elements/thread.
// ---------------------------------------------------------------------------
__global__ __launch_bounds__(256) void moe_split_x(const float4* __restrict__ src,
                                                   uint2* __restrict__ hi,
                                                   uint2* __restrict__ lo) {
  const size_t i = (size_t)blockIdx.x * 256 + threadIdx.x;
  const float4 v = src[i];
  const unsigned short h0 = f2bf(v.x), h1 = f2bf(v.y), h2 = f2bf(v.z), h3 = f2bf(v.w);
  const unsigned short l0 = f2bf(v.x - bf2f(h0)), l1 = f2bf(v.y - bf2f(h1));
  const unsigned short l2 = f2bf(v.z - bf2f(h2)), l3 = f2bf(v.w - bf2f(h3));
  hi[i] = make_uint2(((unsigned)h1 << 16) | h0, ((unsigned)h3 << 16) | h2);
  lo[i] = make_uint2(((unsigned)l1 << 16) | l0, ((unsigned)l3 << 16) | l2);
}

// ---------------------------------------------------------------------------
// Transpose + split: w[e][K][N] fp32 -> hiT/loT[e][N][K] bf16. 32x32 LDS tiles.
// ---------------------------------------------------------------------------
__global__ __launch_bounds__(256) void moe_tsplit_w(const float* __restrict__ w,
                                                    unsigned short* __restrict__ hiT,
                                                    unsigned short* __restrict__ loT,
                                                    int K, int N) {
  __shared__ float sT[32][33];
  const int e  = blockIdx.z;
  const int k0 = blockIdx.y * 32;
  const int n0 = blockIdx.x * 32;
  const int c  = threadIdx.x & 31;
  const int r  = threadIdx.x >> 5;   // 0..7

  const float* src = w + ((size_t)e * K + k0) * N + n0;
#pragma unroll
  for (int i = 0; i < 4; ++i) sT[r + 8 * i][c] = src[(size_t)(r + 8 * i) * N + c];
  __syncthreads();

  const size_t obase = ((size_t)e * N + n0) * (size_t)K + k0;
#pragma unroll
  for (int i = 0; i < 4; ++i) {
    const float v = sT[c][r + 8 * i];          // out[n0+r+8i][k0+c] = in[k0+c][n0+r+8i]
    const unsigned short h = f2bf(v);
    const unsigned short l = f2bf(v - bf2f(h));
    hiT[obase + (size_t)(r + 8 * i) * K + c] = h;
    loT[obase + (size_t)(r + 8 * i) * K + c] = l;
  }
}

// ---------------------------------------------------------------------------
// GEMM1: h = silu(Xg @ Wg) * (Xg @ Wu), bf16x3 split, K = DH.
//   A: gathered x rows via token_of_slot (clamped to row 0; bad rows masked at
//      combine time, so garbage is harmless). Planes xs_hi/xs_lo [T][DH].
//   B: transposed planes wT[e][FF][DH].
// Block: 128 threads = 4 waves; block tile 128(M) x 32(N); wave: 2x2 WMMA tiles.
// Grid: (FF/32=43, CAP/128=10, 8).
// ---------------------------------------------------------------------------
__global__ __launch_bounds__(128) void moe_gemm1(
    const unsigned int* __restrict__ xs_hi, const unsigned int* __restrict__ xs_lo,
    const unsigned int* __restrict__ wgT_hi, const unsigned int* __restrict__ wgT_lo,
    const unsigned int* __restrict__ wuT_hi, const unsigned int* __restrict__ wuT_lo,
    const int* __restrict__ tos,
    unsigned short* __restrict__ hs_hi, unsigned short* __restrict__ hs_lo) {
  __shared__ int sTok[128];

  const int e    = blockIdx.z;
  const int m0   = blockIdx.y * 128;
  const int n0   = blockIdx.x * 32;
  const int tid  = threadIdx.x;
  const int lane = tid & 31;
  const int wm   = (tid >> 5) * 32;

  sTok[tid] = tos[e * CAP + m0 + tid];
  __syncthreads();

  const int mrow = lane & 15;
  const int asel = (lane >= 16) ? 4 : 0;   // A: K-half select (dwords)
  const int bsel = (lane >= 16) ? 4 : 0;   // B: K-half select (dwords)
  const int lh   = lane >> 4;

  // per-lane A row bases (dword units); clamp dropped/empty slots to row 0
  size_t arow[2];
#pragma unroll
  for (int mi = 0; mi < 2; ++mi) {
    int t = sTok[wm + mi * 16 + mrow];
    if (t < 0) t = 0;
    arow[mi] = (size_t)t * (DH / 2);
  }
  // per-lane B row bases (dword units)
  size_t brow[2];
#pragma unroll
  for (int ni = 0; ni < 2; ++ni)
    brow[ni] = ((size_t)e * FF + (n0 + ni * 16 + mrow)) * (size_t)(DH / 2);

  v8f accg[2][2], accu[2][2];
#pragma unroll
  for (int mi = 0; mi < 2; ++mi)
#pragma unroll
    for (int ni = 0; ni < 2; ++ni)
#pragma unroll
      for (int q = 0; q < 8; ++q) { accg[mi][ni][q] = 0.f; accu[mi][ni][q] = 0.f; }

  for (int k0 = 0; k0 < DH / 2; k0 += 16) {   // 16 dwords = 32 bf16 of K; 64 iters
    v16bf ah[2], al[2];
#pragma unroll
    for (int mi = 0; mi < 2; ++mi) {
      ah[mi] = load_a_frag(xs_hi + arow[mi] + k0, asel);
      al[mi] = load_a_frag(xs_lo + arow[mi] + k0, asel);
    }
    {
      v16bf bh[2], bl[2];
#pragma unroll
      for (int ni = 0; ni < 2; ++ni) {
        bh[ni] = load_b_frag(wgT_hi + brow[ni] + k0, bsel);
        bl[ni] = load_b_frag(wgT_lo + brow[ni] + k0, bsel);
      }
#pragma unroll
      for (int mi = 0; mi < 2; ++mi)
#pragma unroll
        for (int ni = 0; ni < 2; ++ni) {
          accg[mi][ni] = wmma_bf16(ah[mi], bh[ni], accg[mi][ni]);
          accg[mi][ni] = wmma_bf16(ah[mi], bl[ni], accg[mi][ni]);
          accg[mi][ni] = wmma_bf16(al[mi], bh[ni], accg[mi][ni]);
        }
    }
    {
      v16bf bh[2], bl[2];
#pragma unroll
      for (int ni = 0; ni < 2; ++ni) {
        bh[ni] = load_b_frag(wuT_hi + brow[ni] + k0, bsel);
        bl[ni] = load_b_frag(wuT_lo + brow[ni] + k0, bsel);
      }
#pragma unroll
      for (int mi = 0; mi < 2; ++mi)
#pragma unroll
        for (int ni = 0; ni < 2; ++ni) {
          accu[mi][ni] = wmma_bf16(ah[mi], bh[ni], accu[mi][ni]);
          accu[mi][ni] = wmma_bf16(ah[mi], bl[ni], accu[mi][ni]);
          accu[mi][ni] = wmma_bf16(al[mi], bh[ni], accu[mi][ni]);
        }
    }
  }

  // epilogue: h = silu(gate) * up, emitted directly as split bf16 planes
#pragma unroll
  for (int mi = 0; mi < 2; ++mi) {
    const size_t rbase = (size_t)e * CAP + m0 + wm + mi * 16 + 8 * lh;
#pragma unroll
    for (int ni = 0; ni < 2; ++ni) {
      const int ncol = n0 + ni * 16 + (lane & 15);
#pragma unroll
      for (int r = 0; r < 8; ++r) {
        const float g = accg[mi][ni][r];
        const float u = accu[mi][ni][r];
        const float s = (g / (1.0f + __expf(-g))) * u;
        const unsigned short hh = f2bf(s);
        const unsigned short ll = f2bf(s - bf2f(hh));
        hs_hi[(rbase + r) * FF + ncol] = hh;
        hs_lo[(rbase + r) * FF + ncol] = ll;
      }
    }
  }
}

// ---------------------------------------------------------------------------
// GEMM2: out = h @ Wd, bf16x3 split, K = FF; scatter with score scaling.
// Grid: (DH/32=64, CAP/128=10, 8).
// ---------------------------------------------------------------------------
__global__ __launch_bounds__(128) void moe_gemm2(
    const unsigned int* __restrict__ hs_hi, const unsigned int* __restrict__ hs_lo,
    const unsigned int* __restrict__ wdT_hi, const unsigned int* __restrict__ wdT_lo,
    const int* __restrict__ tos, const float* __restrict__ scores,
    float* __restrict__ y) {
  __shared__ int sTok[128];

  const int e    = blockIdx.z;
  const int m0   = blockIdx.y * 128;
  const int n0   = blockIdx.x * 32;
  const int tid  = threadIdx.x;
  const int lane = tid & 31;
  const int wm   = (tid >> 5) * 32;

  sTok[tid] = tos[e * CAP + m0 + tid];
  __syncthreads();

  const int mrow = lane & 15;
  const int asel = (lane >= 16) ? 4 : 0;
  const int bsel = (lane >= 16) ? 4 : 0;
  const int lh   = lane >> 4;

  size_t arow[2];
#pragma unroll
  for (int mi = 0; mi < 2; ++mi)
    arow[mi] = ((size_t)e * CAP + m0 + wm + mi * 16 + mrow) * (size_t)(FF / 2);
  size_t brow[2];
#pragma unroll
  for (int ni = 0; ni < 2; ++ni)
    brow[ni] = ((size_t)e * DH + (n0 + ni * 16 + mrow)) * (size_t)(FF / 2);

  v8f acc[2][2];
#pragma unroll
  for (int mi = 0; mi < 2; ++mi)
#pragma unroll
    for (int ni = 0; ni < 2; ++ni)
#pragma unroll
      for (int q = 0; q < 8; ++q) acc[mi][ni][q] = 0.f;

  for (int k0 = 0; k0 < FF / 2; k0 += 16) {   // 43 iterations
    v16bf ah[2], al[2], bh[2], bl[2];
#pragma unroll
    for (int mi = 0; mi < 2; ++mi) {
      ah[mi] = load_a_frag(hs_hi + arow[mi] + k0, asel);
      al[mi] = load_a_frag(hs_lo + arow[mi] + k0, asel);
    }
#pragma unroll
    for (int ni = 0; ni < 2; ++ni) {
      bh[ni] = load_b_frag(wdT_hi + brow[ni] + k0, bsel);
      bl[ni] = load_b_frag(wdT_lo + brow[ni] + k0, bsel);
    }
#pragma unroll
    for (int mi = 0; mi < 2; ++mi)
#pragma unroll
      for (int ni = 0; ni < 2; ++ni) {
        acc[mi][ni] = wmma_bf16(ah[mi], bh[ni], acc[mi][ni]);
        acc[mi][ni] = wmma_bf16(ah[mi], bl[ni], acc[mi][ni]);
        acc[mi][ni] = wmma_bf16(al[mi], bh[ni], acc[mi][ni]);
      }
  }

  // combine: scatter to token rows, scale by gate score; dropped rows skipped
#pragma unroll
  for (int mi = 0; mi < 2; ++mi) {
    const int lrow_base = wm + mi * 16 + 8 * lh;
#pragma unroll
    for (int ni = 0; ni < 2; ++ni) {
      const int ncol = n0 + ni * 16 + (lane & 15);
#pragma unroll
      for (int r = 0; r < 8; ++r) {
        const int t = sTok[lrow_base + r];
        if (t >= 0) {
          y[(size_t)t * DH + ncol] = acc[mi][ni][r] * scores[t] * SCORE_SCALE;
        }
      }
    }
  }
}

// ---------------------------------------------------------------------------
extern "C" void kernel_launch(void* const* d_in, const int* in_sizes, int n_in,
                              void* d_out, int out_size, void* d_ws, size_t ws_size,
                              hipStream_t stream) {
  const float* x      = (const float*)d_in[0];
  const int*   idx    = (const int*)d_in[1];
  const float* scores = (const float*)d_in[2];
  const float* wg     = (const float*)d_in[3];
  const float* wu     = (const float*)d_in[4];
  const float* wd     = (const float*)d_in[5];
  float* y = (float*)d_out;

  // ---- workspace layout (all chunks 256B-multiple) ----
  const size_t SZ_TOS = (size_t)NEXP * CAP * sizeof(int);   // 40960
  const size_t SZ_X   = (size_t)NTOK * DH * 2;              // 32 MiB per plane
  const size_t SZ_W   = (size_t)NEXP * DH * FF * 2;         // 43 MiB per plane
  const size_t SZ_H   = (size_t)NEXP * CAP * FF * 2;        // 26.9 MiB per plane

  char* p = (char*)d_ws;
  int* tos = (int*)p;                          p += SZ_TOS;
  unsigned short* xs_hi  = (unsigned short*)p; p += SZ_X;
  unsigned short* xs_lo  = (unsigned short*)p; p += SZ_X;
  unsigned short* wgT_hi = (unsigned short*)p; p += SZ_W;
  unsigned short* wgT_lo = (unsigned short*)p; p += SZ_W;
  unsigned short* wuT_hi = (unsigned short*)p; p += SZ_W;
  unsigned short* wuT_lo = (unsigned short*)p; p += SZ_W;
  unsigned short* wdT_hi = (unsigned short*)p; p += SZ_W;
  unsigned short* wdT_lo = (unsigned short*)p; p += SZ_W;
  unsigned short* hs_hi  = (unsigned short*)p; p += SZ_H;
  unsigned short* hs_lo  = (unsigned short*)p; p += SZ_H;

  moe_route<<<1, 256, 0, stream>>>(idx, tos);
  moe_split_x<<<(NTOK * DH / 4) / 256, 256, 0, stream>>>(
      (const float4*)x, (uint2*)xs_hi, (uint2*)xs_lo);
  moe_tsplit_w<<<dim3(FF / 32, DH / 32, NEXP), 256, 0, stream>>>(wg, wgT_hi, wgT_lo, DH, FF);
  moe_tsplit_w<<<dim3(FF / 32, DH / 32, NEXP), 256, 0, stream>>>(wu, wuT_hi, wuT_lo, DH, FF);
  moe_tsplit_w<<<dim3(DH / 32, FF / 32, NEXP), 256, 0, stream>>>(wd, wdT_hi, wdT_lo, FF, DH);
  moe_zero_out<<<(NTOK * DH / 4) / 256, 256, 0, stream>>>((float4*)y);

  moe_gemm1<<<dim3(FF / 32, CAP / 128, NEXP), 128, 0, stream>>>(
      (const unsigned int*)xs_hi, (const unsigned int*)xs_lo,
      (const unsigned int*)wgT_hi, (const unsigned int*)wgT_lo,
      (const unsigned int*)wuT_hi, (const unsigned int*)wuT_lo,
      tos, hs_hi, hs_lo);
  moe_gemm2<<<dim3(DH / 32, CAP / 128, NEXP), 128, 0, stream>>>(
      (const unsigned int*)hs_hi, (const unsigned int*)hs_lo,
      (const unsigned int*)wdT_hi, (const unsigned int*)wdT_lo,
      tos, scores, y);
}
